// DiffAttention_22376779612764
// MI455X (gfx1250) — matrix operand order repaired and checked by reading, not compiled, not run on hardware
//
#include <hip/hip_runtime.h>
#include <hip/hip_fp16.h>
#include <hip/hip_bf16.h>

typedef __attribute__((ext_vector_type(16))) _Float16 v16h;
typedef __attribute__((ext_vector_type(8)))  _Float16 h8;
typedef __attribute__((ext_vector_type(8)))  float    v8f;

#define B_   2
#define T_   1024
#define E_   2048
#define H_   16
#define D_   64
#define M_   (B_*T_)                 // 2048
#define LAMBDA_INIT_ 0.78360576653162452f
#define EPS_ 1e-5f

__device__ __forceinline__ v8f wmma16(v16h a, v16h b, v8f c) {
  // D = A(16x32 f16) * B(32x16 f16) + C(16x16 f32)
  return __builtin_amdgcn_wmma_f32_16x16x32_f16(false, a, false, b, (short)0, c, false, false);
}

__device__ __forceinline__ v8f zero8() {
  v8f z;
#pragma unroll
  for (int j = 0; j < 8; ++j) z[j] = 0.0f;
  return z;
}

__device__ __forceinline__ v16h load_afrag(const _Float16* p) {
  // A 16x32 f16 layout: 16 contiguous halves, lane-half offset handled by caller;
  // here: halves [0..7] from p, [8..15] from p+8.
  h8 lo = *(const h8*)p;
  h8 hi8 = *(const h8*)(p + 8);
  v16h a;
#pragma unroll
  for (int j = 0; j < 8; ++j) { a[j] = lo[j]; a[8 + j] = hi8[j]; }
  return a;
}

// ---------------------------------------------------------------- f32 -> f16
__global__ void cvt_f16(const float* __restrict__ in, _Float16* __restrict__ out, int n) {
  int i = blockIdx.x * blockDim.x + threadIdx.x;
  int stride = gridDim.x * blockDim.x;
  for (; i < n; i += stride) out[i] = (_Float16)in[i];
}

// ------------------------------------------------------------------- GEMM
// C[M][N] = A[M][K] (f16, row-major) @ W[N][K]^T (f16, row-major), fp32 out.
// Block: 4 waves (128 threads). Wave: 32(M) x 64(N)  (2 M-tiles share B frags).
// Block tile: 128 x 64. Grid: (N/64, M/128).
__launch_bounds__(128)
__global__ void gemm_f16f32(const _Float16* __restrict__ A, const _Float16* __restrict__ W,
                            float* __restrict__ C, int M, int N, int K) {
  const int lane = threadIdx.x & 31;
  const int wave = threadIdx.x >> 5;
  const int l16  = lane & 15;
  const int hi   = (lane >= 16) ? 1 : 0;
  const int row0 = blockIdx.y * 128 + wave * 32;        // wave owns rows [row0, row0+32)
  const int nbase = blockIdx.x * 64;

  v8f acc[2][4];
#pragma unroll
  for (int mt = 0; mt < 2; ++mt)
#pragma unroll
    for (int nt = 0; nt < 4; ++nt) acc[mt][nt] = zero8();

  const int koffA = hi ? 16 : 0;   // A 16x32: lanes<16 K0..15, else K16..31
  const int koffB = hi ? 16 : 0;   // B 32x16: lanes<16 K0..15, else K16..31
  const _Float16* arow0 = A + (size_t)(row0 + l16) * K;
  const _Float16* arow1 = A + (size_t)(row0 + 16 + l16) * K;

  for (int k0 = 0; k0 < K; k0 += 32) {
    // speculative prefetch of the next K-slab (silently dropped if OOB)
    __builtin_prefetch((const void*)(arow0 + k0 + 256), 0, 1);
    __builtin_prefetch((const void*)(arow1 + k0 + 256), 0, 1);

    v16h a0 = load_afrag(arow0 + k0 + koffA);
    v16h a1 = load_afrag(arow1 + k0 + koffA);
#pragma unroll
    for (int nt = 0; nt < 4; ++nt) {
      const _Float16* wrow = W + (size_t)(nbase + nt * 16 + l16) * K + k0 + koffB;
      __builtin_prefetch((const void*)(wrow + 256), 0, 1);
      v16h b = *(const v16h*)wrow;
      acc[0][nt] = wmma16(a0, b, acc[0][nt]);
      acc[1][nt] = wmma16(a1, b, acc[1][nt]);
    }
  }
  // C layout: VGPR r -> M = r (lanes<16) or r+8 (lanes>=16), N = lane%16
#pragma unroll
  for (int mt = 0; mt < 2; ++mt) {
#pragma unroll
    for (int nt = 0; nt < 4; ++nt) {
#pragma unroll
      for (int r = 0; r < 8; ++r) {
        int rr = row0 + mt * 16 + r + hi * 8;
        int cc = nbase + nt * 16 + l16;
        C[(size_t)rr * N + cc] = acc[mt][nt][r];
      }
    }
  }
}

// ------------------------------------------------- RoPE / rel_pos prep for Q,K
// q_pre,k_pre: fp32 [M][E] with e = head*64 + d (32 heads x 64)
// Qh,Kh: f16 [b][32][T][64]; Q scaled by 1/sqrt(D)
__global__ void prep_qk(const float* __restrict__ qpre, const float* __restrict__ kpre,
                        const float* __restrict__ relpos,
                        _Float16* __restrict__ Qh, _Float16* __restrict__ Kh) {
  int idx = blockIdx.x * blockDim.x + threadIdx.x;   // M_*32*32 threads
  if (idx >= M_ * 32 * 32) return;
  int d2   = idx & 31;          // pair index 0..31
  int head = (idx >> 5) & 31;
  int m    = idx >> 10;
  int t    = m & (T_ - 1);
  int b    = m >> 10;

  float fr  = powf(10000.0f, -((float)(2 * d2)) / (float)D_);
  float ang = (float)t * fr;
  float c = cosf(ang), s = sinf(ang);

  size_t src = (size_t)m * E_ + (size_t)head * D_ + 2 * d2;
  float q1 = qpre[src], q2 = qpre[src + 1];
  size_t dst = ((size_t)(b * 32 + head) * T_ + t) * D_ + 2 * d2;
  Qh[dst]     = (_Float16)((q1 * c - q2 * s) * 0.125f);
  Qh[dst + 1] = (_Float16)((q1 * s + q2 * c) * 0.125f);

  float r1 = relpos[t * D_ + 2 * d2], r2 = relpos[t * D_ + 2 * d2 + 1];
  float k1 = kpre[src] * r1, k2 = kpre[src + 1] * r2;
  Kh[dst]     = (_Float16)(k1 * c - k2 * s);
  Kh[dst + 1] = (_Float16)(k1 * s + k2 * c);
}

// ------------------------------------------- V transpose: [M][E] -> [b][16][128][T]
__global__ void prep_v(const float* __restrict__ vpre, _Float16* __restrict__ Vth) {
  int idx = blockIdx.x * blockDim.x + threadIdx.x;   // M_*E_ threads
  if (idx >= M_ * E_) return;
  int e = idx & (E_ - 1);
  int m = idx >> 11;
  int t = m & (T_ - 1);
  int b = m >> 10;
  int h = e >> 7;
  int d = e & 127;
  Vth[((size_t)(b * H_ + h) * 128 + d) * T_ + t] = (_Float16)vpre[idx];
}

// ------------------------------------------------------------ lambda scalar
__global__ void lam_kernel(const float* __restrict__ lq1, const float* __restrict__ lq2,
                           const float* __restrict__ lk1, const float* __restrict__ lk2,
                           float* __restrict__ lam) {
  int i = threadIdx.x;   // 32 threads
  float s1 = lq1[i] * lk1[i] + lq1[i + 32] * lk1[i + 32];
  float s2 = lq2[i] * lk2[i] + lq2[i + 32] * lk2[i + 32];
#pragma unroll
  for (int mk = 16; mk >= 1; mk >>= 1) {
    s1 += __shfl_xor(s1, mk, 32);
    s2 += __shfl_xor(s2, mk, 32);
  }
  if (i == 0) *lam = expf(s1) - expf(s2) + LAMBDA_INIT_;
}

// ---------------------------------------- fused differential flash attention
// grid (T/64, H, B); 4 waves/block; wave owns 16 q-rows for both sub-heads.
__launch_bounds__(128)
__global__ void flash_diff(const _Float16* __restrict__ Qh, const _Float16* __restrict__ Kh,
                           const _Float16* __restrict__ Vth, const float* __restrict__ lam_p,
                           const float* __restrict__ sw, _Float16* __restrict__ attnh) {
  __shared__ _Float16 lds_p[4][16][72];   // per-wave P staging (padded rows)

  const int lane  = threadIdx.x & 31;
  const int wave  = threadIdx.x >> 5;
  const int l16   = lane & 15;
  const int hi    = (lane >= 16) ? 1 : 0;
  const int b     = blockIdx.z;
  const int h     = blockIdx.y;
  const int qtile = blockIdx.x * 64;
  const int qrow0 = qtile + wave * 16;

  // Preload Q A-fragments for both sub-heads, both K-steps (d 0..31, 32..63)
  v16h aq[2][2];
#pragma unroll
  for (int sub = 0; sub < 2; ++sub) {
    const _Float16* qbase = Qh + ((size_t)(b * 32 + 2 * h + sub) * T_ + (qrow0 + l16)) * D_;
#pragma unroll
    for (int ks = 0; ks < 2; ++ks) {
      int ko = ks * 32 + (hi ? 16 : 0);
      aq[sub][ks] = load_afrag(qbase + ko);
    }
  }

  v8f o[2][8];
  float mrow[2][8], lrow[2][8];
#pragma unroll
  for (int sub = 0; sub < 2; ++sub) {
#pragma unroll
    for (int nt = 0; nt < 8; ++nt) o[sub][nt] = zero8();
#pragma unroll
    for (int r = 0; r < 8; ++r) { mrow[sub][r] = -1e30f; lrow[sub][r] = 0.0f; }
  }

  _Float16* lp = &lds_p[wave][0][0];
  const int kend = qtile + 64;                    // causal upper bound (uniform per block)

  for (int kb = 0; kb < kend; kb += 64) {
#pragma unroll
    for (int sub = 0; sub < 2; ++sub) {
      const _Float16* kmat = Kh + (size_t)(b * 32 + 2 * h + sub) * T_ * D_;
      // S tile = Q K^T  (16 x 64)
      v8f s[4];
#pragma unroll
      for (int nt = 0; nt < 4; ++nt) s[nt] = zero8();
#pragma unroll
      for (int ks = 0; ks < 2; ++ks) {
#pragma unroll
        for (int nt = 0; nt < 4; ++nt) {
          const _Float16* kp = kmat + (size_t)(kb + nt * 16 + l16) * D_ + ks * 32 + (hi ? 16 : 0);
          v16h bk = *(const v16h*)kp;
          s[nt] = wmma16(aq[sub][ks], bk, s[nt]);
        }
      }
      // causal mask
#pragma unroll
      for (int nt = 0; nt < 4; ++nt) {
        int kt = kb + nt * 16 + l16;
#pragma unroll
        for (int r = 0; r < 8; ++r) {
          int qt = qrow0 + r + hi * 8;
          s[nt][r] = (kt > qt) ? -1e9f : s[nt][r];
        }
      }
      // online softmax (row reductions within 16-lane halves via xor shuffles)
      float tmax[8], corr[8], tsum[8];
#pragma unroll
      for (int r = 0; r < 8; ++r) {
        float mx = fmaxf(fmaxf(s[0][r], s[1][r]), fmaxf(s[2][r], s[3][r]));
#pragma unroll
        for (int mk = 1; mk < 16; mk <<= 1) mx = fmaxf(mx, __shfl_xor(mx, mk, 32));
        tmax[r] = mx;
      }
#pragma unroll
      for (int r = 0; r < 8; ++r) {
        float mn = fmaxf(mrow[sub][r], tmax[r]);
        corr[r] = expf(mrow[sub][r] - mn);
        mrow[sub][r] = mn;
        tsum[r] = 0.0f;
      }
#pragma unroll
      for (int nt = 0; nt < 4; ++nt) {
#pragma unroll
        for (int r = 0; r < 8; ++r) {
          float p = expf(s[nt][r] - mrow[sub][r]);
          s[nt][r] = p;
          tsum[r] += p;
        }
      }
#pragma unroll
      for (int r = 0; r < 8; ++r) {
#pragma unroll
        for (int mk = 1; mk < 16; mk <<= 1) tsum[r] += __shfl_xor(tsum[r], mk, 32);
        lrow[sub][r] = lrow[sub][r] * corr[r] + tsum[r];
      }
#pragma unroll
      for (int nt = 0; nt < 8; ++nt)
#pragma unroll
        for (int r = 0; r < 8; ++r) o[sub][nt][r] *= corr[r];

      // Stage P (C-layout) into LDS, re-read in A-layout
#pragma unroll
      for (int nt = 0; nt < 4; ++nt)
#pragma unroll
        for (int r = 0; r < 8; ++r)
          lp[(r + hi * 8) * 72 + nt * 16 + l16] = (_Float16)s[nt][r];

      __asm__ volatile("s_wait_dscnt 0" ::: "memory");

      // O += P @ V   (V B-fragment contiguous in t thanks to transposed Vth)
#pragma unroll
      for (int ks = 0; ks < 2; ++ks) {
        v16h ap;
        int ko = ks * 32 + (hi ? 16 : 0);
#pragma unroll
        for (int j = 0; j < 8; ++j) {
          ap[j]     = lp[l16 * 72 + ko + j];
          ap[8 + j] = lp[l16 * 72 + ko + 8 + j];
        }
#pragma unroll
        for (int nt = 0; nt < 8; ++nt) {
          const _Float16* vp = Vth + ((size_t)(b * H_ + h) * 128 + nt * 16 + l16) * T_
                                   + kb + ks * 32 + (hi ? 16 : 0);
          v16h bv = *(const v16h*)vp;
          o[sub][nt] = wmma16(ap, bv, o[sub][nt]);
        }
      }
      __asm__ volatile("s_wait_dscnt 0" ::: "memory");
    }
  }

  // Epilogue: O = O1/l1 - lam*O2/l2, RMS-norm over 128, subln scale, f16 out
  const float lam = *lam_p;
  float ssq[8];
#pragma unroll
  for (int r = 0; r < 8; ++r) ssq[r] = 0.0f;
#pragma unroll
  for (int nt = 0; nt < 8; ++nt) {
#pragma unroll
    for (int r = 0; r < 8; ++r) {
      float val = o[0][nt][r] / lrow[0][r] - lam * (o[1][nt][r] / lrow[1][r]);
      o[0][nt][r] = val;
      ssq[r] += val * val;
    }
  }
  float scale[8];
#pragma unroll
  for (int r = 0; r < 8; ++r) {
#pragma unroll
    for (int mk = 1; mk < 16; mk <<= 1) ssq[r] += __shfl_xor(ssq[r], mk, 32);
    scale[r] = rsqrtf(ssq[r] * (1.0f / 128.0f) + EPS_);
  }
#pragma unroll
  for (int nt = 0; nt < 8; ++nt) {
    float w = sw[nt * 16 + l16];
#pragma unroll
    for (int r = 0; r < 8; ++r) {
      int trow = qrow0 + r + hi * 8;
      attnh[((size_t)b * T_ + trow) * E_ + h * 128 + nt * 16 + l16] =
          (_Float16)(o[0][nt][r] * scale[r] * w);
    }
  }
}

// ----------------------------------------------------------------- launcher
extern "C" void kernel_launch(void* const* d_in, const int* in_sizes, int n_in,
                              void* d_out, int out_size, void* d_ws, size_t ws_size,
                              hipStream_t stream) {
  const float* x   = (const float*)d_in[0];
  const float* rel = (const float*)d_in[1];
  const float* wq  = (const float*)d_in[2];
  const float* wk  = (const float*)d_in[3];
  const float* wv  = (const float*)d_in[4];
  const float* lq1 = (const float*)d_in[5];
  const float* lq2 = (const float*)d_in[6];
  const float* lk1 = (const float*)d_in[7];
  const float* lk2 = (const float*)d_in[8];
  const float* sw  = (const float*)d_in[9];
  const float* wo  = (const float*)d_in[10];
  float* out = (float*)d_out;

  char* ws = (char*)d_ws;
  const size_t SZH = (size_t)M_ * E_ * sizeof(_Float16);  // 8 MB
  const size_t SZF = (size_t)M_ * E_ * sizeof(float);     // 16 MB
  _Float16* xh    = (_Float16*)(ws);
  _Float16* wh    = (_Float16*)(ws + SZH);
  float*    q_pre = (float*)(ws + 2 * SZH);
  float*    k_pre = (float*)(ws + 2 * SZH + SZF);
  float*    v_pre = (float*)(ws + 2 * SZH + 2 * SZF);
  _Float16* Qh    = (_Float16*)(ws + 2 * SZH + 3 * SZF);
  _Float16* Kh    = (_Float16*)(ws + 3 * SZH + 3 * SZF);
  _Float16* Vth   = (_Float16*)(ws + 4 * SZH + 3 * SZF);
  _Float16* attnh = (_Float16*)q_pre;     // alias: q_pre dead after prep_qk
  float*    lamp  = (float*)k_pre;        // alias: k_pre dead after prep_qk

  const int nME = M_ * E_;        // 4,194,304 (x and each weight)
  dim3 cb(256), cg(4096);
  dim3 gb(128), gg(E_ / 64, M_ / 128);    // GEMM block tile 128x64

  cvt_f16<<<cg, cb, 0, stream>>>(x, xh, nME);

  cvt_f16<<<cg, cb, 0, stream>>>(wq, wh, nME);
  gemm_f16f32<<<gg, gb, 0, stream>>>(xh, wh, q_pre, M_, E_, E_);
  cvt_f16<<<cg, cb, 0, stream>>>(wk, wh, nME);
  gemm_f16f32<<<gg, gb, 0, stream>>>(xh, wh, k_pre, M_, E_, E_);
  cvt_f16<<<cg, cb, 0, stream>>>(wv, wh, nME);
  gemm_f16f32<<<gg, gb, 0, stream>>>(xh, wh, v_pre, M_, E_, E_);

  prep_qk<<<(M_ * 32 * 32) / 256, 256, 0, stream>>>(q_pre, k_pre, rel, Qh, Kh);
  prep_v<<<(M_ * E_) / 256, 256, 0, stream>>>(v_pre, Vth);
  lam_kernel<<<1, 32, 0, stream>>>(lq1, lq2, lk1, lk2, lamp);

  flash_diff<<<dim3(T_ / 64, H_, B_), 128, 0, stream>>>(Qh, Kh, Vth, lamp, sw, attnh);

  cvt_f16<<<cg, cb, 0, stream>>>(wo, wh, nME);
  gemm_f16f32<<<gg, gb, 0, stream>>>(attnh, wh, out, M_, E_, E_);
}
